// ResnetBlockMoE2D_2800318677420
// MI455X (gfx1250) — compile-verified, matrix operand order checked by reading, not run on hardware
//
#include <hip/hip_runtime.h>
#include <hip/hip_bf16.h>
#include <stdint.h>

// ---------------- problem constants ----------------
#define BATCH   4
#define CCH     256
#define HH      64
#define WWID    64
#define HP      66                    // padded H/W for conv halo
#define NTOK    (BATCH*HH*WWID)       // 16384 tokens / output pixels
#define NGROUPS 32
#define EPSV    1e-5f
#define NEXP    4
#define IDIM    1024
#define KCONV   (9*CCH)               // 2304

#define LDSPITCH 40                   // bf16 elements per LDS row (32 data + 8 pad)
#define MT      256                   // workgroup M tile (32 rows per wave)

typedef __attribute__((ext_vector_type(16))) __bf16 v16bf;
typedef __attribute__((ext_vector_type(8)))  float  v8f;
typedef __attribute__((ext_vector_type(4)))  int    v4i;

union FragU { v16bf v; uint4 q[2]; };

// Per-lane WMMA fragment load from LDS: 16 bf16 = two 16-byte chunks at +0/+16
// elements (16-bit 16x32 fragment layout: lane<16 holds K {0..7,16..23},
// lane>=16 holds K {8..15,24..31}; caller adds hi*8 to the base).
__device__ __forceinline__ v16bf load_frag(const __bf16* p) {
    FragU f;
    f.q[0] = *(const uint4*)(p);
    f.q[1] = *(const uint4*)(p + 16);
    return f.v;
}

__device__ __forceinline__ void zero_acc(v8f* a, int n) {
    for (int t = 0; t < n; ++t)
        for (int j = 0; j < 8; ++j) a[t][j] = 0.f;
}

// ---- CDNA5 async copy global -> LDS (16B per lane), tracked by ASYNCcnt ----
__device__ __forceinline__ void async_cp16(void* lds_dst, const void* gsrc) {
#if __has_builtin(__builtin_amdgcn_global_load_async_to_lds_b128)
    typedef __attribute__((address_space(1))) v4i gv4i_t;   // global int4
    typedef __attribute__((address_space(3))) v4i lv4i_t;   // LDS int4
    __builtin_amdgcn_global_load_async_to_lds_b128(
        (gv4i_t*)(uintptr_t)gsrc,
        (lv4i_t*)(uint32_t)(uintptr_t)lds_dst,
        0, 0);
#else
    asm volatile("global_load_async_to_lds_b128 %0, %1, off"
                 :: "v"((uint32_t)(uintptr_t)lds_dst), "v"(gsrc)
                 : "memory");
#endif
}

__device__ __forceinline__ void async_wait0() {
#if __has_builtin(__builtin_amdgcn_s_wait_asynccnt)
    __builtin_amdgcn_s_wait_asynccnt(0);
#else
    asm volatile("s_wait_asynccnt 0" ::: "memory");
#endif
}

// ---------------- utility kernels ----------------
__global__ void zero_u32_kernel(uint32_t* __restrict__ p, int n) {
    int i = blockIdx.x * blockDim.x + threadIdx.x;
    if (i < n) p[i] = 0u;
}

__global__ void f32_to_bf16_kernel(const float* __restrict__ s, __bf16* __restrict__ d, int n) {
    int i = blockIdx.x * blockDim.x + threadIdx.x;
    if (i < n) d[i] = (__bf16)s[i];
}

// conv weight OIHW [O,C,3,3] -> [O, K=tap*256+c] bf16 (K fastest, matches im2col)
__global__ void pack_convw_kernel(const float* __restrict__ w, __bf16* __restrict__ wpk) {
    int idx = blockIdx.x * blockDim.x + threadIdx.x;
    if (idx >= CCH * KCONV) return;
    int o = idx / KCONV, k = idx - o * KCONV;
    int tap = k >> 8, c = k & 255;
    int dy = tap / 3, dx = tap - dy * 3;
    wpk[idx] = (__bf16)w[((size_t)(o * CCH + c) * 3 + dy) * 3 + dx];
}

// ---------------- GroupNorm (+optional swish), NCHW f32 -> padded NHWC bf16 ----------------
__global__ void __launch_bounds__(256)
groupnorm_kernel(const float* __restrict__ x, const float* __restrict__ gamma,
                 const float* __restrict__ beta, __bf16* __restrict__ out,
                 int pad, int do_swish) {
    __shared__ float ssum[256], ssq[256];
    int bg = blockIdx.x;               // B*NGROUPS blocks
    int b = bg >> 5, gi = bg & 31;
    int c0 = gi * 8;
    int tid = threadIdx.x;
    const float* xb = x + (size_t)b * CCH * 4096;

    float s = 0.f, sq = 0.f;
    for (int i = tid; i < 8 * 4096; i += 256) {
        int c = c0 + (i >> 12), sp = i & 4095;
        float v = xb[(size_t)c * 4096 + sp];
        s += v; sq += v * v;
    }
    ssum[tid] = s; ssq[tid] = sq;
    __syncthreads();
    for (int off = 128; off > 0; off >>= 1) {
        if (tid < off) { ssum[tid] += ssum[tid + off]; ssq[tid] += ssq[tid + off]; }
        __syncthreads();
    }
    float mean = ssum[0] * (1.f / 32768.f);
    float var  = ssq[0] * (1.f / 32768.f) - mean * mean;
    float rstd = rsqrtf(var + EPSV);

    int Wp = WWID + 2 * pad;
    for (int i = tid; i < 8 * 4096; i += 256) {
        int c = c0 + (i >> 12), sp = i & 4095;
        float v = xb[(size_t)c * 4096 + sp];
        float y = (v - mean) * rstd * gamma[c] + beta[c];
        if (do_swish) y = y / (1.f + __expf(-y));
        int yy = sp >> 6, xx = sp & 63;
        size_t oi = (((size_t)b * Wp + (yy + pad)) * Wp + (xx + pad)) * CCH + c;
        out[oi] = (__bf16)y;
    }
}

// ---------------- conv3x3 as implicit GEMM with WMMA bf16 + async LDS pipeline ----------
// M=16384 pixels, N=256 out-ch, K=2304. grid(M/256, N/64); 8 waves, 32x64 per wave.
__global__ void __launch_bounds__(256)
conv3x3_wmma_kernel(const __bf16* __restrict__ actpad, const __bf16* __restrict__ wpk,
                    const float* __restrict__ bias, const float* __restrict__ residual,
                    float* __restrict__ outNCHW) {
    __shared__ __align__(16) __bf16 As[2][MT * LDSPITCH];
    __shared__ __align__(16) __bf16 Bs[2][64 * LDSPITCH];

    int tid = threadIdx.x;
    int wid = tid >> 5, lane = tid & 31;
    int l16 = lane & 15, hi = lane >> 4;
    int mbase = blockIdx.x * MT;
    int n0 = blockIdx.y * 64;
    int m0 = mbase + wid * 32;

    // Per-thread A-copy assignments: four (row, chunk) pairs; pixel decode loop-invariant.
    const __bf16* apix[4];
    int arow[4], ach[4];
#pragma unroll
    for (int q = 0; q < 4; ++q) {
        int p = tid + q * 256;               // 0..1023
        int r = p >> 2, ch = p & 3;
        int pr = mbase + r;
        int pb = pr >> 12, sp = pr & 4095;
        int py = sp >> 6, px = sp & 63;
        apix[q] = actpad + (((size_t)pb * HP + py) * HP + px) * CCH + ch * 8;
        arow[q] = r; ach[q] = ch;
    }
    int bcol = tid >> 2, bch = tid & 3;      // B-copy assignment (one chunk/thread)

    auto issue = [&](int k0, int s) {
        int tap = k0 >> 8;
        int dy = tap / 3, dx = tap - dy * 3;
        int cbase = k0 & 255;
        size_t aoff = ((size_t)dy * HP + dx) * CCH + cbase;
#pragma unroll
        for (int q = 0; q < 4; ++q)
            async_cp16(&As[s][arow[q] * LDSPITCH + ach[q] * 8], apix[q] + aoff);
        async_cp16(&Bs[s][bcol * LDSPITCH + bch * 8],
                   wpk + (size_t)(n0 + bcol) * KCONV + k0 + bch * 8);
    };

    v8f acc[2][4]; zero_acc(&acc[0][0], 8);
    issue(0, 0);
    async_wait0(); __syncthreads();
    int cur = 0;
    for (int k0 = 0; k0 < KCONV; k0 += 32) {
        if (k0 + 32 < KCONV) issue(k0 + 32, cur ^ 1);
        // Batch all fragment loads, then stream the 8 WMMAs back-to-back.
        v16bf afrag[2], bfrag[4];
#pragma unroll
        for (int u = 0; u < 2; ++u)
            afrag[u] = load_frag(&As[cur][(wid * 32 + u * 16 + l16) * LDSPITCH + hi * 8]);
#pragma unroll
        for (int t = 0; t < 4; ++t)
            bfrag[t] = load_frag(&Bs[cur][(t * 16 + l16) * LDSPITCH + hi * 8]);
#pragma unroll
        for (int u = 0; u < 2; ++u)
#pragma unroll
            for (int t = 0; t < 4; ++t)
                acc[u][t] = __builtin_amdgcn_wmma_f32_16x16x32_bf16(
                    false, afrag[u], false, bfrag[t], (short)0, acc[u][t], false, false);
        async_wait0(); __syncthreads();
        cur ^= 1;
    }

#pragma unroll
    for (int u = 0; u < 2; ++u)
#pragma unroll
    for (int t = 0; t < 4; ++t) {
        int col = n0 + t * 16 + l16;
        float bv = bias[col];
#pragma unroll
        for (int j = 0; j < 8; ++j) {
            int r = m0 + u * 16 + hi * 8 + j;
            int rb = r >> 12, rsp = r & 4095;
            size_t oi = ((size_t)rb * CCH + col) * 4096 + rsp;
            float v = acc[u][t][j] + bv;
            if (residual) v += residual[oi];
            outNCHW[oi] = v;
        }
    }
}

// ---------------- generic GEMM: C[M,N] f32 = A[M,K]bf16 * B[N,K]bf16^T ----------------
// grid(M/256, N/64); 32x64 per wave; optional per-row scale + accumulate epilogue.
__global__ void __launch_bounds__(256)
gemm_bf16_kernel(const __bf16* __restrict__ A, const __bf16* __restrict__ Bw,
                 float* __restrict__ C, int M, int N, int K,
                 const float* __restrict__ rowscale, int rs_stride, int accumulate) {
    __shared__ __align__(16) __bf16 As[2][MT * LDSPITCH];
    __shared__ __align__(16) __bf16 Bs[2][64 * LDSPITCH];

    int tid = threadIdx.x;
    int wid = tid >> 5, lane = tid & 31;
    int l16 = lane & 15, hi = lane >> 4;
    int mbase = blockIdx.x * MT;
    int n0 = blockIdx.y * 64;
    int m0 = mbase + wid * 32;

    int bcol = tid >> 2, bch = tid & 3;

    auto issue = [&](int k0, int s) {
#pragma unroll
        for (int q = 0; q < 4; ++q) {
            int p = tid + q * 256;
            int r = p >> 2, ch = p & 3;
            async_cp16(&As[s][r * LDSPITCH + ch * 8],
                       A + (size_t)(mbase + r) * K + k0 + ch * 8);
        }
        async_cp16(&Bs[s][bcol * LDSPITCH + bch * 8],
                   Bw + (size_t)(n0 + bcol) * K + k0 + bch * 8);
    };

    v8f acc[2][4]; zero_acc(&acc[0][0], 8);
    issue(0, 0);
    async_wait0(); __syncthreads();
    int cur = 0;
    for (int k0 = 0; k0 < K; k0 += 32) {
        if (k0 + 32 < K) issue(k0 + 32, cur ^ 1);
        v16bf afrag[2], bfrag[4];
#pragma unroll
        for (int u = 0; u < 2; ++u)
            afrag[u] = load_frag(&As[cur][(wid * 32 + u * 16 + l16) * LDSPITCH + hi * 8]);
#pragma unroll
        for (int t = 0; t < 4; ++t)
            bfrag[t] = load_frag(&Bs[cur][(t * 16 + l16) * LDSPITCH + hi * 8]);
#pragma unroll
        for (int u = 0; u < 2; ++u)
#pragma unroll
            for (int t = 0; t < 4; ++t)
                acc[u][t] = __builtin_amdgcn_wmma_f32_16x16x32_bf16(
                    false, afrag[u], false, bfrag[t], (short)0, acc[u][t], false, false);
        async_wait0(); __syncthreads();
        cur ^= 1;
    }

#pragma unroll
    for (int u = 0; u < 2; ++u)
#pragma unroll
    for (int t = 0; t < 4; ++t) {
        int col = n0 + t * 16 + l16;
#pragma unroll
        for (int j = 0; j < 8; ++j) {
            int r = m0 + u * 16 + hi * 8 + j;
            float sc = rowscale ? rowscale[(size_t)r * rs_stride] : 1.f;
            float v = sc * acc[u][t][j];
            size_t oi = (size_t)r * N + col;
            if (accumulate) C[oi] += v; else C[oi] = v;
        }
    }
}

// ---------------- fused gate+up projection with SiLU(g)*u epilogue -> bf16 --------------
// grid(M/256, N/32); per wave 32x32 against both weight matrices (8 WMMAs/step).
__global__ void __launch_bounds__(256)
gateup_bf16_kernel(const __bf16* __restrict__ A, const __bf16* __restrict__ Gw,
                   const __bf16* __restrict__ Uw, __bf16* __restrict__ Y,
                   int M, int N, int K) {
    __shared__ __align__(16) __bf16 As[2][MT * LDSPITCH];
    __shared__ __align__(16) __bf16 Bs[2][64 * LDSPITCH];

    int tid = threadIdx.x;
    int wid = tid >> 5, lane = tid & 31;
    int l16 = lane & 15, hi = lane >> 4;
    int mbase = blockIdx.x * MT;
    int n0 = blockIdx.y * 32;
    int m0 = mbase + wid * 32;

    int bcol = tid >> 2, bch = tid & 3;      // bcol 0..63: <32 -> G, >=32 -> U
    const __bf16* bsrc_row = (bcol < 32)
        ? Gw + (size_t)(n0 + bcol) * K
        : Uw + (size_t)(n0 + bcol - 32) * K;

    auto issue = [&](int k0, int s) {
#pragma unroll
        for (int q = 0; q < 4; ++q) {
            int p = tid + q * 256;
            int r = p >> 2, ch = p & 3;
            async_cp16(&As[s][r * LDSPITCH + ch * 8],
                       A + (size_t)(mbase + r) * K + k0 + ch * 8);
        }
        async_cp16(&Bs[s][bcol * LDSPITCH + bch * 8], bsrc_row + k0 + bch * 8);
    };

    v8f accg[2][2], accu[2][2];
    zero_acc(&accg[0][0], 4); zero_acc(&accu[0][0], 4);
    issue(0, 0);
    async_wait0(); __syncthreads();
    int cur = 0;
    for (int k0 = 0; k0 < K; k0 += 32) {
        if (k0 + 32 < K) issue(k0 + 32, cur ^ 1);
        v16bf afrag[2], bg[2], bu[2];
#pragma unroll
        for (int u = 0; u < 2; ++u)
            afrag[u] = load_frag(&As[cur][(wid * 32 + u * 16 + l16) * LDSPITCH + hi * 8]);
#pragma unroll
        for (int t = 0; t < 2; ++t) {
            bg[t] = load_frag(&Bs[cur][(t * 16 + l16) * LDSPITCH + hi * 8]);
            bu[t] = load_frag(&Bs[cur][(32 + t * 16 + l16) * LDSPITCH + hi * 8]);
        }
#pragma unroll
        for (int u = 0; u < 2; ++u)
#pragma unroll
            for (int t = 0; t < 2; ++t) {
                accg[u][t] = __builtin_amdgcn_wmma_f32_16x16x32_bf16(
                    false, afrag[u], false, bg[t], (short)0, accg[u][t], false, false);
                accu[u][t] = __builtin_amdgcn_wmma_f32_16x16x32_bf16(
                    false, afrag[u], false, bu[t], (short)0, accu[u][t], false, false);
            }
        async_wait0(); __syncthreads();
        cur ^= 1;
    }

#pragma unroll
    for (int u = 0; u < 2; ++u)
#pragma unroll
    for (int t = 0; t < 2; ++t) {
        int col = n0 + t * 16 + l16;
#pragma unroll
        for (int j = 0; j < 8; ++j) {
            int r = m0 + u * 16 + hi * 8 + j;
            float g = accg[u][t][j], uu = accu[u][t][j];
            float y = (g / (1.f + __expf(-g))) * uu;   // silu(g)*u
            Y[(size_t)r * N + col] = (__bf16)y;
        }
    }
}

// ---------------- router: logits -> softmax -> top-1 dense combine weights -------------
__global__ void router_kernel(const __bf16* __restrict__ tok, const float* __restrict__ gw,
                              float* __restrict__ wfull) {
    int m = blockIdx.x * blockDim.x + threadIdx.x;
    if (m >= NTOK) return;
    const __bf16* t = tok + (size_t)m * CCH;
    float l0 = 0.f, l1 = 0.f, l2 = 0.f, l3 = 0.f;
    for (int c = 0; c < CCH; ++c) {
        float v = (float)t[c];
        l0 += v * gw[c];
        l1 += v * gw[CCH + c];
        l2 += v * gw[2 * CCH + c];
        l3 += v * gw[3 * CCH + c];
    }
    float mx = fmaxf(fmaxf(l0, l1), fmaxf(l2, l3));
    float e0 = __expf(l0 - mx), e1 = __expf(l1 - mx);
    float e2 = __expf(l2 - mx), e3 = __expf(l3 - mx);
    float inv = 1.f / (e0 + e1 + e2 + e3);
    float p[4] = { e0 * inv, e1 * inv, e2 * inv, e3 * inv };
    int best = 0; float bp = p[0];
    for (int e = 1; e < 4; ++e) if (p[e] > bp) { bp = p[e]; best = e; }
    for (int e = 0; e < 4; ++e) wfull[(size_t)m * 4 + e] = (e == best) ? bp : 0.f;
}

// ---------------- final: out(NCHW) = h(NCHW) + routed(tokens-major) ----------------
__global__ void final_add_kernel(const float* __restrict__ h, const float* __restrict__ routed,
                                 float* __restrict__ out) {
    int i = blockIdx.x * blockDim.x + threadIdx.x;
    if (i >= BATCH * CCH * 4096) return;
    int b = i >> 20;                 // C*H*W = 2^20
    int c = (i >> 12) & 255;
    int sp = i & 4095;
    int m = (b << 12) + sp;
    out[i] = h[i] + routed[(size_t)m * CCH + c];
}

// ---------------- host orchestration ----------------
extern "C" void kernel_launch(void* const* d_in, const int* in_sizes, int n_in,
                              void* d_out, int out_size, void* d_ws, size_t ws_size,
                              hipStream_t stream) {
    const float* x       = (const float*)d_in[0];
    const float* gn1_g   = (const float*)d_in[1];
    const float* gn1_b   = (const float*)d_in[2];
    const float* conv1_w = (const float*)d_in[3];
    const float* conv1_b = (const float*)d_in[4];
    const float* gn2_g   = (const float*)d_in[5];
    const float* gn2_b   = (const float*)d_in[6];
    const float* conv2_w = (const float*)d_in[7];
    const float* conv2_b = (const float*)d_in[8];
    const float* gnm_g   = (const float*)d_in[9];
    const float* gnm_b   = (const float*)d_in[10];
    const float* gate_w  = (const float*)d_in[11];
    const float* exp_gw  = (const float*)d_in[12];
    const float* exp_uw  = (const float*)d_in[13];
    const float* exp_dw  = (const float*)d_in[14];
    const float* sh_gw   = (const float*)d_in[15];
    const float* sh_uw   = (const float*)d_in[16];
    const float* sh_dw   = (const float*)d_in[17];
    float* out = (float*)d_out;

    char* ws = (char*)d_ws;
    size_t off = 0;
    auto alloc = [&](size_t bytes) -> void* {
        void* p = ws + off;
        off = (off + bytes + 255) & ~(size_t)255;
        return p;
    };

    __bf16* actpad = (__bf16*)alloc((size_t)BATCH * HP * HP * CCH * 2);
    float*  tmp1   = (float*) alloc((size_t)BATCH * CCH * 4096 * 4);
    float*  hbuf   = (float*) alloc((size_t)BATCH * CCH * 4096 * 4);
    __bf16* tok    = (__bf16*)alloc((size_t)NTOK * CCH * 2);
    float*  wfull  = (float*) alloc((size_t)NTOK * 4 * 4);
    __bf16* Y      = (__bf16*)alloc((size_t)NTOK * IDIM * 2);
    float*  routed = (float*) alloc((size_t)NTOK * CCH * 4);
    __bf16* wpk1   = (__bf16*)alloc((size_t)CCH * KCONV * 2);
    __bf16* wpk2   = (__bf16*)alloc((size_t)CCH * KCONV * 2);
    __bf16* bgw    = (__bf16*)alloc((size_t)NEXP * IDIM * CCH * 2);
    __bf16* buw    = (__bf16*)alloc((size_t)NEXP * IDIM * CCH * 2);
    __bf16* bdw    = (__bf16*)alloc((size_t)NEXP * CCH * IDIM * 2);
    __bf16* bsgw   = (__bf16*)alloc((size_t)IDIM * CCH * 2);
    __bf16* bsuw   = (__bf16*)alloc((size_t)IDIM * CCH * 2);
    __bf16* bsdw   = (__bf16*)alloc((size_t)CCH * IDIM * 2);
    (void)ws_size; (void)in_sizes; (void)n_in; (void)out_size;

    // Zero padded activation buffer (halo must be 0 every call; reused for both convs).
    {
        int n32 = (BATCH * HP * HP * CCH) / 2;
        zero_u32_kernel<<<(n32 + 255) / 256, 256, 0, stream>>>((uint32_t*)actpad, n32);
    }

    // Weight conversion / packing (re-done each call for determinism).
    auto cvt = [&](const float* s, __bf16* d, int n) {
        f32_to_bf16_kernel<<<(n + 255) / 256, 256, 0, stream>>>(s, d, n);
    };
    pack_convw_kernel<<<(CCH * KCONV + 255) / 256, 256, 0, stream>>>(conv1_w, wpk1);
    pack_convw_kernel<<<(CCH * KCONV + 255) / 256, 256, 0, stream>>>(conv2_w, wpk2);
    cvt(exp_gw, bgw, NEXP * IDIM * CCH);
    cvt(exp_uw, buw, NEXP * IDIM * CCH);
    cvt(exp_dw, bdw, NEXP * CCH * IDIM);
    cvt(sh_gw, bsgw, IDIM * CCH);
    cvt(sh_uw, bsuw, IDIM * CCH);
    cvt(sh_dw, bsdw, CCH * IDIM);

    // ResnetBlock
    groupnorm_kernel<<<BATCH * NGROUPS, 256, 0, stream>>>(x, gn1_g, gn1_b, actpad, 1, 1);
    conv3x3_wmma_kernel<<<dim3(NTOK / MT, CCH / 64), 256, 0, stream>>>(
        actpad, wpk1, conv1_b, nullptr, tmp1);
    groupnorm_kernel<<<BATCH * NGROUPS, 256, 0, stream>>>(tmp1, gn2_g, gn2_b, actpad, 1, 1);
    conv3x3_wmma_kernel<<<dim3(NTOK / MT, CCH / 64), 256, 0, stream>>>(
        actpad, wpk2, conv2_b, x, hbuf);   // residual add: h = x + conv2(...)

    // MoE pre-norm (no swish) -> tokens [NTOK, C] bf16
    groupnorm_kernel<<<BATCH * NGROUPS, 256, 0, stream>>>(hbuf, gnm_g, gnm_b, tok, 0, 0);

    // Router
    router_kernel<<<NTOK / 256, 256, 0, stream>>>(tok, gate_w, wfull);

    // Routed experts (dense compute, top-1 weight as row-scale; expert 0 initializes)
    for (int e = 0; e < NEXP; ++e) {
        gateup_bf16_kernel<<<dim3(NTOK / MT, IDIM / 32), 256, 0, stream>>>(
            tok, bgw + (size_t)e * IDIM * CCH, buw + (size_t)e * IDIM * CCH,
            Y, NTOK, IDIM, CCH);
        gemm_bf16_kernel<<<dim3(NTOK / MT, CCH / 64), 256, 0, stream>>>(
            Y, bdw + (size_t)e * CCH * IDIM, routed, NTOK, CCH, IDIM,
            wfull + e, 4, e > 0 ? 1 : 0);
    }
    // Shared expert (scale 1, accumulate)
    gateup_bf16_kernel<<<dim3(NTOK / MT, IDIM / 32), 256, 0, stream>>>(
        tok, bsgw, bsuw, Y, NTOK, IDIM, CCH);
    gemm_bf16_kernel<<<dim3(NTOK / MT, CCH / 64), 256, 0, stream>>>(
        Y, bsdw, routed, NTOK, CCH, IDIM, nullptr, 0, 1);

    // out = moe_out + h
    final_add_kernel<<<(BATCH * CCH * 4096 + 255) / 256, 256, 0, stream>>>(hbuf, routed, out);
}
